// MaskedSelfAttention_28200755266002
// MI455X (gfx1250) — matrix operand order repaired and checked by reading, not compiled, not run on hardware
//
#include <hip/hip_runtime.h>

// Problem constants (B,T,C,H fixed by the reference)
#define BB 4
#define TT 2048
#define CC 1024
#define HH 16
#define DD 64

typedef __attribute__((ext_vector_type(16))) _Float16 v16h;
typedef __attribute__((ext_vector_type(8)))  float    v8f;
typedef __attribute__((ext_vector_type(4)))  int      v4i;

union FragAB {
    int4 q[2];     // 32 bytes
    v16h h;
};

__device__ __forceinline__ v8f wmma_f16(const FragAB& a, const FragAB& b, v8f c) {
    return __builtin_amdgcn_wmma_f32_16x16x32_f16(
        /*neg_a=*/false, a.h, /*neg_b=*/false, b.h,
        /*c_mod=*/(short)0, c, /*reuse_a=*/false, /*reuse_b=*/false);
}

// ---------------------------------------------------------------------------
// Async global->LDS staging (CDNA5 GLOBAL_LOAD_ASYNC_TO_LDS_B128, ASYNCcnt).
// Builtin takes (v4i addrspace(1)*, v4i addrspace(3)*, imm offset, imm cpol).
// ---------------------------------------------------------------------------
typedef __attribute__((address_space(1))) v4i* gv4i_ptr;
typedef __attribute__((address_space(3))) v4i* lv4i_ptr;

#if defined(__HIP_DEVICE_COMPILE__) && __has_builtin(__builtin_amdgcn_global_load_async_to_lds_b128)
#define HAVE_ASYNC_LDS 1
__device__ __forceinline__ void async_cp16(const void* g, void* l) {
    __builtin_amdgcn_global_load_async_to_lds_b128((gv4i_ptr)g, (lv4i_ptr)l, 0, 0);
}
#if __has_builtin(__builtin_amdgcn_s_wait_asynccnt)
#define ASYNC_WAIT_LE(n) __builtin_amdgcn_s_wait_asynccnt(n)
#else
#define ASYNC_WAIT_LE(n) asm volatile("s_wait_asynccnt %0" ::"n"(n) : "memory")
#endif
#else
#define HAVE_ASYNC_LDS 0
__device__ __forceinline__ void async_cp16(const void* g, void* l) {
    *(int4*)l = *(const int4*)g;
}
#define ASYNC_WAIT_LE(n)
#endif

// ---------------------------------------------------------------------------
// fp32 -> f16 elementwise convert
// ---------------------------------------------------------------------------
__global__ __launch_bounds__(256) void cvt_f32_f16(const float* __restrict__ src,
                                                   _Float16* __restrict__ dst, int n) {
    int i = blockIdx.x * 256 + threadIdx.x;
    if (i < n) dst[i] = (_Float16)src[i];
}

// ---------------------------------------------------------------------------
// 1024x1024 fp32 -> f16 transpose (Wt[n][k] = W[k][n]), 32x32 LDS tiles
// ---------------------------------------------------------------------------
__global__ __launch_bounds__(256) void transpose_cvt(const float* __restrict__ W,
                                                     _Float16* __restrict__ Wt) {
    __shared__ float tile[32][33];
    int tx = threadIdx.x & 31;
    int ty4 = threadIdx.x >> 5;           // 0..7
    int bx = blockIdx.x, by = blockIdx.y; // 32 x 32 tiles
#pragma unroll
    for (int p = 0; p < 4; ++p) {
        int ty = ty4 + p * 8;
        tile[ty][tx] = W[(size_t)(by * 32 + ty) * 1024 + bx * 32 + tx];
    }
    __syncthreads();
#pragma unroll
    for (int p = 0; p < 4; ++p) {
        int ty = ty4 + p * 8;
        Wt[(size_t)(bx * 32 + ty) * 1024 + by * 32 + tx] = (_Float16)tile[tx][ty];
    }
}

// ---------------------------------------------------------------------------
// GEMM: C[8192,1024] = A[8192,1024](f16) * W (via Wt[n][k], f16) + bias
//   EPI==0 : out f16 to [B,H,T,D] layout, value scaled by `scale`
//   EPI==1 : out fp32 row-major [8192,1024]
// Tile 128x128 per 256-thread (8-wave) WG; each wave computes 32x64; K-step 32.
// Double-buffered LDS: async loads of tile k+1 overlap WMMA on tile k.
// ---------------------------------------------------------------------------
#define GTILE 10240           // one 128x80B tile
#define GBUF  (2 * GTILE)     // A + B tiles for one stage

template <int EPI>
__global__ __launch_bounds__(256) void gemm_k(const _Float16* __restrict__ A,
                                              const _Float16* __restrict__ Bt,
                                              const float* __restrict__ bias,
                                              void* __restrict__ out, float scale) {
    __shared__ __align__(16) unsigned char lds[2 * GBUF];

    const int tid = threadIdx.x;
    const int wave = tid >> 5, lane = tid & 31;
    const int halfk = lane >> 4, l15 = lane & 15;
    const int wm = wave & 3, wn = wave >> 2;          // 4 x 2 wave grid
    const int m0 = blockIdx.x * 128, n0 = blockIdx.y * 128;
    const int ldrow = tid >> 1, ldpart = tid & 1;     // 128 rows, 32B per thread

    const _Float16* gA = A + (size_t)(m0 + ldrow) * 1024 + ldpart * 16;
    const _Float16* gB = Bt + (size_t)(n0 + ldrow) * 1024 + ldpart * 16;
    const int ldoff = ldrow * 80 + ldpart * 32;

    v8f acc[2][4] = {};

    // prologue: stage tile 0 into buffer 0 (4 async b128 per thread)
    {
        unsigned char* la = lds + ldoff;
        unsigned char* lb = lds + GTILE + ldoff;
        async_cp16(gA, la);      async_cp16(gA + 8, la + 16);
        async_cp16(gB, lb);      async_cp16(gB + 8, lb + 16);
    }

    int buf = 0;
    for (int k0 = 0; k0 < 1024; k0 += 32, buf ^= 1) {
        // issue next tile into the other buffer (overlaps with this tile's math)
        if (k0 + 32 < 1024) {
            unsigned char* la = lds + (buf ^ 1) * GBUF + ldoff;
            unsigned char* lb = lds + (buf ^ 1) * GBUF + GTILE + ldoff;
            const _Float16* ga = gA + k0 + 32;
            const _Float16* gb = gB + k0 + 32;
            async_cp16(ga, la);      async_cp16(ga + 8, la + 16);
            async_cp16(gb, lb);      async_cp16(gb + 8, lb + 16);
            ASYNC_WAIT_LE(4);        // current tile's 4 copies retired (in-order)
        } else {
            ASYNC_WAIT_LE(0);
        }
        __syncthreads();             // all waves' copies for this tile visible

        unsigned char* As = lds + buf * GBUF;
        unsigned char* Bs = As + GTILE;

        FragAB af[2];
#pragma unroll
        for (int a = 0; a < 2; ++a) {
            int row = wm * 32 + a * 16 + l15;
            af[a].q[0] = *(const int4*)(As + row * 80 + halfk * 16);
            af[a].q[1] = *(const int4*)(As + row * 80 + halfk * 16 + 32);
        }
#pragma unroll
        for (int f = 0; f < 4; ++f) {
            FragAB bf;
            int col = wn * 64 + (f << 4) + l15;
            bf.q[0] = *(const int4*)(Bs + col * 80 + halfk * 16);
            bf.q[1] = *(const int4*)(Bs + col * 80 + halfk * 16 + 32);
#pragma unroll
            for (int a = 0; a < 2; ++a)
                acc[a][f] = wmma_f16(af[a], bf, acc[a][f]);
        }
        __syncthreads();             // readers done before buffer is re-issued
    }

    // epilogue
#pragma unroll
    for (int f = 0; f < 4; ++f) {
        int c = n0 + wn * 64 + (f << 4) + l15;
        float bv = bias[c];
#pragma unroll
        for (int a = 0; a < 2; ++a) {
            int mrow = m0 + wm * 32 + a * 16;
#pragma unroll
            for (int i = 0; i < 8; ++i) {
                int r = mrow + i + halfk * 8;      // C layout: M = i (+8 upper half)
                float v = acc[a][f][i] + bv;
                if (EPI == 0) {
                    v *= scale;
                    int b = r >> 11, t = r & (TT - 1);
                    int h = c >> 6, d = c & (DD - 1);
                    ((_Float16*)out)[((((size_t)b * HH + h) * TT + t) << 6) + d] = (_Float16)v;
                } else {
                    ((float*)out)[(size_t)r * 1024 + c] = v;
                }
            }
        }
    }
}

// ---------------------------------------------------------------------------
// Flash attention: per (bh, 64-row q tile). 128 threads / 4 waves; each wave
// owns 16 q rows. Streaming softmax over 64-key tiles, causal.
// ---------------------------------------------------------------------------
__global__ __launch_bounds__(128) void flash_k(const _Float16* __restrict__ Q,
                                               const _Float16* __restrict__ K,
                                               const _Float16* __restrict__ V,
                                               _Float16* __restrict__ Y) {
    __shared__ __align__(16) unsigned char sQ[64 * 144];
    __shared__ __align__(16) unsigned char sK[64 * 144];   // [key][d]
    __shared__ __align__(16) unsigned char sV[64 * 144];   // transposed: [d][key]
    __shared__ __align__(16) unsigned char sP[64 * 144];   // 4 waves x 16 rows

    const int tid = threadIdx.x;
    const int wave = tid >> 5, lane = tid & 31;
    const int halfk = lane >> 4, l15 = lane & 15;
    const int qt = blockIdx.x;          // 0..31
    const int bh = blockIdx.y;          // 0..63
    const size_t base = (size_t)bh * TT * DD;
    const int q0 = qt * 64;
    const int ldrow = tid >> 1, ldpart = tid & 1;

    // load Q tile (64 rows x 128B, 2 threads/row -> 64B each) via async copies
    {
        const _Float16* g = Q + base + (size_t)(q0 + ldrow) * DD + ldpart * 32;
        unsigned char* p = sQ + ldrow * 144 + ldpart * 64;
#pragma unroll
        for (int u = 0; u < 4; ++u) async_cp16(g + u * 8, p + u * 16);
    }
    ASYNC_WAIT_LE(0);
    __syncthreads();

    // preload Q fragments (reused for every key tile)
    FragAB qf[2];
    {
        int row = (wave << 4) + l15;
#pragma unroll
        for (int ks = 0; ks < 2; ++ks) {
            qf[ks].q[0] = *(const int4*)(sQ + row * 144 + ks * 64 + halfk * 16);
            qf[ks].q[1] = *(const int4*)(sQ + row * 144 + ks * 64 + halfk * 16 + 32);
        }
    }

    float mrun[8], lrun[8];
    v8f oacc[4] = {};
#pragma unroll
    for (int i = 0; i < 8; ++i) { mrun[i] = -3.0e38f; lrun[i] = 0.0f; }

    for (int j = 0; j <= qt; ++j) {
        __syncthreads();  // previous iteration finished reading sK/sV
        // K tile: [key][d] rows (async global->LDS)
        {
            const _Float16* g = K + base + (size_t)(j * 64 + ldrow) * DD + ldpart * 32;
            unsigned char* p = sK + ldrow * 144 + ldpart * 64;
#pragma unroll
            for (int u = 0; u < 4; ++u) async_cp16(g + u * 8, p + u * 16);
        }
        // V tile transposed into [d][key]
#pragma unroll
        for (int idx = tid; idx < 64 * 32; idx += 128) {
            int key = idx >> 5, dp = idx & 31;
            unsigned int w2 = *(const unsigned int*)(V + base + (size_t)(j * 64 + key) * DD + dp * 2);
            *(unsigned short*)(sV + (dp * 2 + 0) * 144 + key * 2) = (unsigned short)(w2 & 0xffffu);
            *(unsigned short*)(sV + (dp * 2 + 1) * 144 + key * 2) = (unsigned short)(w2 >> 16);
        }
        ASYNC_WAIT_LE(0);
        __syncthreads();

        // S = Q * K^T   (1/sqrt(D) scale already folded into Q)
        v8f s[4] = {};
#pragma unroll
        for (int ks = 0; ks < 2; ++ks) {
#pragma unroll
            for (int f = 0; f < 4; ++f) {
                FragAB bf;
                int col = (f << 4) + l15;    // key index in tile
                bf.q[0] = *(const int4*)(sK + col * 144 + ks * 64 + halfk * 16);
                bf.q[1] = *(const int4*)(sK + col * 144 + ks * 64 + halfk * 16 + 32);
                s[f] = wmma_f16(qf[ks], bf, s[f]);
            }
        }

        // causal mask only on the diagonal tile
        if (j == qt) {
#pragma unroll
            for (int f = 0; f < 4; ++f) {
                int n = (f << 4) + l15;
#pragma unroll
                for (int i = 0; i < 8; ++i) {
                    int m = (wave << 4) + i + halfk * 8;
                    if (n > m) s[f][i] = -3.0e38f;
                }
            }
        }

        // online softmax (rows live within 16-lane halves -> xor 1,2,4,8)
        float mnew[8], scl[8];
#pragma unroll
        for (int i = 0; i < 8; ++i) {
            float v = fmaxf(fmaxf(s[0][i], s[1][i]), fmaxf(s[2][i], s[3][i]));
            v = fmaxf(v, __shfl_xor(v, 1));
            v = fmaxf(v, __shfl_xor(v, 2));
            v = fmaxf(v, __shfl_xor(v, 4));
            v = fmaxf(v, __shfl_xor(v, 8));
            mnew[i] = fmaxf(mrun[i], v);
            scl[i] = __expf(mrun[i] - mnew[i]);
            mrun[i] = mnew[i];
        }

        // P = exp(S - mnew); stash to per-wave LDS as f16 (A-fragment source)
        unsigned char* pbase = sP + (wave << 4) * 144;
#pragma unroll
        for (int f = 0; f < 4; ++f) {
            int n = (f << 4) + l15;
#pragma unroll
            for (int i = 0; i < 8; ++i) {
                float p = __expf(s[f][i] - mnew[i]);
                s[f][i] = p;
                int m = i + halfk * 8;
                *(_Float16*)(pbase + m * 144 + n * 2) = (_Float16)p;
            }
        }
#pragma unroll
        for (int i = 0; i < 8; ++i) {
            float rs = (s[0][i] + s[1][i]) + (s[2][i] + s[3][i]);
            rs += __shfl_xor(rs, 1);
            rs += __shfl_xor(rs, 2);
            rs += __shfl_xor(rs, 4);
            rs += __shfl_xor(rs, 8);
            lrun[i] = lrun[i] * scl[i] + rs;
#pragma unroll
            for (int f = 0; f < 4; ++f) oacc[f][i] *= scl[i];
        }

        // O += P * V
#pragma unroll
        for (int ks = 0; ks < 2; ++ks) {
            FragAB pf;
            pf.q[0] = *(const int4*)(pbase + l15 * 144 + ks * 64 + halfk * 16);
            pf.q[1] = *(const int4*)(pbase + l15 * 144 + ks * 64 + halfk * 16 + 32);
#pragma unroll
            for (int f = 0; f < 4; ++f) {
                FragAB bf;
                int col = (f << 4) + l15;    // d index
                bf.q[0] = *(const int4*)(sV + col * 144 + ks * 64 + halfk * 16);
                bf.q[1] = *(const int4*)(sV + col * 144 + ks * 64 + halfk * 16 + 32);
                oacc[f] = wmma_f16(pf, bf, oacc[f]);
            }
        }
    }

    // finalize: O/l  ->  y f16 [B,T,C], c = h*64 + d
    {
        const int b = bh >> 4, h = bh & (HH - 1);
        const int tq = q0 + (wave << 4);
#pragma unroll
        for (int i = 0; i < 8; ++i) {
            float inv = 1.0f / lrun[i];
            int t = tq + i + halfk * 8;
#pragma unroll
            for (int f = 0; f < 4; ++f) {
                int d = (f << 4) + l15;
                Y[((size_t)(b * TT + t) << 10) + (h << 6) + d] = (_Float16)(oacc[f][i] * inv);
            }
        }
    }
}

// ---------------------------------------------------------------------------
// host launcher
// ---------------------------------------------------------------------------
extern "C" void kernel_launch(void* const* d_in, const int* in_sizes, int n_in,
                              void* d_out, int out_size, void* d_ws, size_t ws_size,
                              hipStream_t stream) {
    const float* x  = (const float*)d_in[0];
    const float* Wq = (const float*)d_in[1];
    const float* bq = (const float*)d_in[2];
    const float* Wk = (const float*)d_in[3];
    const float* bk = (const float*)d_in[4];
    const float* Wv = (const float*)d_in[5];
    const float* bv = (const float*)d_in[6];
    const float* Wp = (const float*)d_in[7];
    const float* bp = (const float*)d_in[8];

    char* ws = (char*)d_ws;
    const size_t nX = (size_t)BB * TT * CC;          // 8M elements
    _Float16* xh  = (_Float16*)(ws);                            // 16 MB
    _Float16* wqt = (_Float16*)(ws + 16777216);                 // 2 MB each
    _Float16* wkt = (_Float16*)(ws + 16777216 + 2097152);
    _Float16* wvt = (_Float16*)(ws + 16777216 + 2 * 2097152);
    _Float16* wpt = (_Float16*)(ws + 16777216 + 3 * 2097152);
    _Float16* qh  = (_Float16*)(ws + 25165824);                 // [B,H,T,D] 16 MB
    _Float16* kh  = (_Float16*)(ws + 25165824 + 16777216);
    _Float16* vh  = (_Float16*)(ws + 25165824 + 2 * 16777216);
    _Float16* yh  = (_Float16*)(ws + 25165824 + 3 * 16777216);  // [B,T,C] 16 MB

    cvt_f32_f16<<<(int)(nX / 256), 256, 0, stream>>>(x, xh, (int)nX);
    transpose_cvt<<<dim3(32, 32), 256, 0, stream>>>(Wq, wqt);
    transpose_cvt<<<dim3(32, 32), 256, 0, stream>>>(Wk, wkt);
    transpose_cvt<<<dim3(32, 32), 256, 0, stream>>>(Wv, wvt);
    transpose_cvt<<<dim3(32, 32), 256, 0, stream>>>(Wp, wpt);

    dim3 ggrid(64, 8);     // M/128 x N/128
    gemm_k<0><<<ggrid, 256, 0, stream>>>(xh, wqt, bq, (void*)qh, 0.125f);  // 1/sqrt(64)
    gemm_k<0><<<ggrid, 256, 0, stream>>>(xh, wkt, bk, (void*)kh, 1.0f);
    gemm_k<0><<<ggrid, 256, 0, stream>>>(xh, wvt, bv, (void*)vh, 1.0f);

    flash_k<<<dim3(32, BB * HH), 128, 0, stream>>>(qh, kh, vh, yh);

    gemm_k<1><<<ggrid, 256, 0, stream>>>(yh, wpt, bp, d_out, 1.0f);
}
